// LowRankMultiheadAttention_56255481643695
// MI455X (gfx1250) — matrix operand order repaired and checked by reading, not compiled, not run on hardware
//
#include <hip/hip_runtime.h>

typedef __attribute__((ext_vector_type(16))) __bf16 v16bf;
typedef __attribute__((ext_vector_type(8)))  __bf16 v8bf;
typedef __attribute__((ext_vector_type(4)))  __bf16 v4bf;
typedef __attribute__((ext_vector_type(8)))  float  v8f;
typedef __attribute__((ext_vector_type(4)))  float  v4f;

#define E_DIM 1024
#define H_NUM 16
#define R_DIM 256
#define HD_DIM 64
#define B_DIM 2
#define S_DIM 2048
#define M_DIM (B_DIM * S_DIM)   // 4096

static __device__ __forceinline__ v16bf pack16(v8bf lo, v8bf hi) {
  v16bf r;
#pragma unroll
  for (int i = 0; i < 8; ++i) { r[i] = lo[i]; r[i + 8] = hi[i]; }
  return r;
}

static __device__ __forceinline__ v8f zero8() {
  v8f r;
#pragma unroll
  for (int i = 0; i < 8; ++i) r[i] = 0.0f;
  return r;
}

static __device__ __forceinline__ v4bf cvt4(v4f a) {
  v4bf r;
#pragma unroll
  for (int i = 0; i < 4; ++i) r[i] = (__bf16)a[i];
  return r;
}

// ---------------------------------------------------------------------------
// Tiled GEMM: C(4096xN) = A(4096xK,f32) @ W(KxN,f32) + bias, scaled.
// N, K, output mode are compile-time -> constant-stride addressing, no
// epilogue branches. Block = 256 threads (8 waves), tile 128x128, K-step 32.
// Each wave computes 32x64 = 2x4 wmma accumulators.
// ---------------------------------------------------------------------------
template <int N, int K, bool WRITE_F32>
__global__ __launch_bounds__(256) void lr_gemm_kernel(
    const float* __restrict__ A, const float* __restrict__ W,
    const float* __restrict__ bias, float* __restrict__ outF,
    __bf16* __restrict__ outB, float scale)
{
  __shared__ __align__(16) __bf16 As[128 * 40];   // A tile, k-major, pad 32->40
  __shared__ __align__(16) __bf16 Bt[128 * 40];   // W tile, n-major (transposed)
  const int t      = threadIdx.x;
  const int lane   = t & 31;
  const int wid    = t >> 5;
  const int lane16 = lane & 15;
  const int half   = lane >> 4;
  const int wm     = wid >> 1;                    // 0..3
  const int wn     = wid & 1;                     // 0..1
  const int mb     = blockIdx.x * 128;
  const int nb     = blockIdx.y * 128;

  // B-tile 4x4 register-transpose assignment
  const int kg = t >> 5;                          // 0..7  -> k rows 4*kg..+3
  const int ng = t & 31;                          // 0..31 -> n cols 4*ng..+3

  v8f acc[2][4];
#pragma unroll
  for (int mi = 0; mi < 2; ++mi)
#pragma unroll
    for (int ni = 0; ni < 4; ++ni) acc[mi][ni] = zero8();

  for (int k0 = 0; k0 < K; k0 += 32) {
    // A tile: 128x32, float4 loads -> packed b64 LDS stores
#pragma unroll
    for (int i = 0; i < 4; ++i) {
      int slot = t + i * 256;                     // 1024 float4 slots
      int r = slot >> 3, kc = (slot & 7) * 4;
      v4f a = *(const v4f*)&A[(size_t)(mb + r) * K + (k0 + kc)];
      *(v4bf*)&As[r * 40 + kc] = cvt4(a);
    }
    // W tile: 32x128, 4x4 block transpose in registers
    {
      v4f w0 = *(const v4f*)&W[(size_t)(k0 + 4 * kg + 0) * N + nb + 4 * ng];
      v4f w1 = *(const v4f*)&W[(size_t)(k0 + 4 * kg + 1) * N + nb + 4 * ng];
      v4f w2 = *(const v4f*)&W[(size_t)(k0 + 4 * kg + 2) * N + nb + 4 * ng];
      v4f w3 = *(const v4f*)&W[(size_t)(k0 + 4 * kg + 3) * N + nb + 4 * ng];
#pragma unroll
      for (int c = 0; c < 4; ++c) {
        v4bf col;
        col[0] = (__bf16)w0[c]; col[1] = (__bf16)w1[c];
        col[2] = (__bf16)w2[c]; col[3] = (__bf16)w3[c];
        *(v4bf*)&Bt[(4 * ng + c) * 40 + 4 * kg] = col;
      }
    }
    __syncthreads();

    v16bf afr[2], bfr[4];
#pragma unroll
    for (int mi = 0; mi < 2; ++mi) {
      const __bf16* p = &As[(wm * 32 + mi * 16 + lane16) * 40 + half * 8];
      afr[mi] = pack16(*(const v8bf*)p, *(const v8bf*)(p + 16));
    }
#pragma unroll
    for (int ni = 0; ni < 4; ++ni) {
      const __bf16* p = &Bt[(wn * 64 + ni * 16 + lane16) * 40 + half * 8];
      bfr[ni] = pack16(*(const v8bf*)p, *(const v8bf*)(p + 16));
    }
#pragma unroll
    for (int mi = 0; mi < 2; ++mi)
#pragma unroll
      for (int ni = 0; ni < 4; ++ni)
        acc[mi][ni] = __builtin_amdgcn_wmma_f32_16x16x32_bf16(
            false, afr[mi], false, bfr[ni], (short)0, acc[mi][ni], false, false);
    __syncthreads();
  }

#pragma unroll
  for (int mi = 0; mi < 2; ++mi)
#pragma unroll
    for (int ni = 0; ni < 4; ++ni) {
      const int col  = nb + wn * 64 + ni * 16 + lane16;
      const int row0 = mb + wm * 32 + mi * 16 + half * 8;
      const float bv = bias[col];
      if constexpr (WRITE_F32) {
        float* p = outF + (size_t)row0 * N + col;
#pragma unroll
        for (int v = 0; v < 8; ++v) p[(size_t)v * N] = (acc[mi][ni][v] + bv) * scale;
      } else {
        __bf16* p = outB + (size_t)row0 * N + col;
#pragma unroll
        for (int v = 0; v < 8; ++v) p[(size_t)v * N] = (__bf16)((acc[mi][ni][v] + bv) * scale);
      }
    }
}

// ---------------------------------------------------------------------------
// Transpose V (B,S,E) bf16 -> Vt (B*H, HD, S) bf16 for contiguous PV B-frags.
// ---------------------------------------------------------------------------
__global__ __launch_bounds__(256) void lr_transpose_v(
    const __bf16* __restrict__ v, __bf16* __restrict__ vt)
{
  size_t idx = (size_t)blockIdx.x * 256 + threadIdx.x;  // output index
  int s = (int)(idx & (S_DIM - 1));
  size_t r = idx >> 11;
  int d = (int)(r & (HD_DIM - 1)); r >>= 6;
  int h = (int)(r & (H_NUM - 1));
  int b = (int)(r >> 4);
  vt[idx] = v[((size_t)(b * S_DIM + s)) * E_DIM + h * HD_DIM + d];
}

// ---------------------------------------------------------------------------
// Flash attention: block = 4 waves, 64 query rows of one (b,h).
// Wave owns 16 q rows; softmax stats are wave-local (C-layout rows live in
// lanes 0-15 / 16-31 halves -> shfl_xor<16 reductions).
// ---------------------------------------------------------------------------
__global__ __launch_bounds__(128) void lr_flash_attn(
    const __bf16* __restrict__ qb, const __bf16* __restrict__ kb,
    const __bf16* __restrict__ vt, float* __restrict__ attn)
{
  __shared__ __align__(16) __bf16 Plds[4][16][64];   // per-wave P tile
  const int qt = blockIdx.x;
  const int bh = blockIdx.y;
  const int b  = bh >> 4;
  const int h  = bh & 15;
  const int t      = threadIdx.x;
  const int lane   = t & 31;
  const int wid    = t >> 5;
  const int lane16 = lane & 15;
  const int half   = lane >> 4;
  const int qrow0  = qt * 64 + wid * 16;

  v16bf aq[2];                                       // Q A-frags (HD=64 -> 2 k-chunks)
  {
    const __bf16* qp = qb + ((size_t)(b * S_DIM + qrow0 + lane16)) * E_DIM + h * HD_DIM;
#pragma unroll
    for (int c = 0; c < 2; ++c) {
      const __bf16* p = qp + c * 32 + half * 8;
      aq[c] = pack16(*(const v8bf*)p, *(const v8bf*)(p + 16));
    }
  }

  float m_run[8], l_run[8];
  v8f o_acc[4];
#pragma unroll
  for (int ni = 0; ni < 4; ++ni) o_acc[ni] = zero8();
#pragma unroll
  for (int v = 0; v < 8; ++v) { m_run[v] = -3.0e38f; l_run[v] = 0.0f; }

  for (int kt = 0; kt < S_DIM / 64; ++kt) {
    // scores tile: 16(q) x 64(keys); 1/sqrt(HD) already folded into q
    v8f s_acc[4];
#pragma unroll
    for (int ni = 0; ni < 4; ++ni) {
      s_acc[ni] = zero8();
#pragma unroll
      for (int c = 0; c < 2; ++c) {
        const __bf16* p = kb + ((size_t)(b * S_DIM + kt * 64 + ni * 16 + lane16)) * E_DIM
                             + h * HD_DIM + c * 32 + half * 8;
        v16bf bk = pack16(*(const v8bf*)p, *(const v8bf*)(p + 16));
        s_acc[ni] = __builtin_amdgcn_wmma_f32_16x16x32_bf16(
            false, aq[c], false, bk, (short)0, s_acc[ni], false, false);
      }
    }
    // online softmax over this 64-key tile (wave-local)
#pragma unroll
    for (int v = 0; v < 8; ++v) {
      float mx = s_acc[0][v];
#pragma unroll
      for (int ni = 1; ni < 4; ++ni) mx = fmaxf(mx, s_acc[ni][v]);
#pragma unroll
      for (int off = 1; off < 16; off <<= 1) mx = fmaxf(mx, __shfl_xor(mx, off));
      const float mnew = fmaxf(m_run[v], mx);
      const float corr = __expf(m_run[v] - mnew);
      m_run[v] = mnew;
      float rs = 0.0f;
#pragma unroll
      for (int ni = 0; ni < 4; ++ni) {
        const float e = __expf(s_acc[ni][v] - mnew);
        rs += e;
        Plds[wid][v + half * 8][ni * 16 + lane16] = (__bf16)e;  // C-layout -> LDS
      }
#pragma unroll
      for (int off = 1; off < 16; off <<= 1) rs += __shfl_xor(rs, off);
      l_run[v] = l_run[v] * corr + rs;
#pragma unroll
      for (int ni = 0; ni < 4; ++ni) o_acc[ni][v] *= corr;
    }
    asm volatile("s_wait_dscnt 0" ::: "memory");     // P stores visible to reads
    v16bf ap[2];                                     // P back as A-frags
#pragma unroll
    for (int c = 0; c < 2; ++c) {
      const __bf16* p = &Plds[wid][lane16][c * 32 + half * 8];
      ap[c] = pack16(*(const v8bf*)p, *(const v8bf*)(p + 16));
    }
#pragma unroll
    for (int ni = 0; ni < 4; ++ni) {
#pragma unroll
      for (int c = 0; c < 2; ++c) {
        const __bf16* p = vt + ((size_t)bh * HD_DIM + ni * 16 + lane16) * S_DIM
                             + kt * 64 + c * 32 + half * 8;
        v16bf bv = pack16(*(const v8bf*)p, *(const v8bf*)(p + 16));
        o_acc[ni] = __builtin_amdgcn_wmma_f32_16x16x32_bf16(
            false, ap[c], false, bv, (short)0, o_acc[ni], false, false);
      }
    }
  }
#pragma unroll
  for (int v = 0; v < 8; ++v) {
    const float inv = 1.0f / l_run[v];
    const int row = qrow0 + v + half * 8;
    float* p = attn + ((size_t)(b * S_DIM + row)) * E_DIM + h * HD_DIM;
#pragma unroll
    for (int ni = 0; ni < 4; ++ni) p[ni * 16 + lane16] = o_acc[ni][v] * inv;
  }
}

extern "C" void kernel_launch(void* const* d_in, const int* in_sizes, int n_in,
                              void* d_out, int out_size, void* d_ws, size_t ws_size,
                              hipStream_t stream) {
  (void)in_sizes; (void)n_in; (void)out_size; (void)ws_size;
  const float* query = (const float*)d_in[0];
  const float* key_  = (const float*)d_in[1];
  const float* value = (const float*)d_in[2];
  const float* Wq_lo = (const float*)d_in[3];
  const float* bq_lo = (const float*)d_in[4];
  const float* Wq_hi = (const float*)d_in[5];
  const float* bq_hi = (const float*)d_in[6];
  const float* Wk_lo = (const float*)d_in[7];
  const float* bk_lo = (const float*)d_in[8];
  const float* Wk_hi = (const float*)d_in[9];
  const float* bk_hi = (const float*)d_in[10];
  const float* Wv_lo = (const float*)d_in[11];
  const float* bv_lo = (const float*)d_in[12];
  const float* Wv_hi = (const float*)d_in[13];
  const float* bv_hi = (const float*)d_in[14];
  const float* Wo_lo = (const float*)d_in[15];
  const float* bo_lo = (const float*)d_in[16];
  const float* Wo_hi = (const float*)d_in[17];
  const float* bo_hi = (const float*)d_in[18];

  char* ws = (char*)d_ws;
  float*  tmp  = (float*)(ws);                        //  4 MB (4096x256 f32)
  __bf16* qbuf = (__bf16*)(ws + (4u  << 20));         //  8 MB
  __bf16* kbuf = (__bf16*)(ws + (12u << 20));         //  8 MB
  __bf16* vbuf = (__bf16*)(ws + (20u << 20));         //  8 MB
  __bf16* vtb  = (__bf16*)(ws + (28u << 20));         //  8 MB
  float*  attn = (float*)(ws + (36u << 20));          // 16 MB

  dim3 gLo(M_DIM / 128, R_DIM / 128);                 // (32, 2)
  dim3 gHi(M_DIM / 128, E_DIM / 128);                 // (32, 8)

  // q/k/v low-rank projections (scale 1/sqrt(HD)=0.125 folded into q)
  lr_gemm_kernel<R_DIM, E_DIM, true ><<<gLo, 256, 0, stream>>>(query, Wq_lo, bq_lo, tmp, nullptr, 1.0f);
  lr_gemm_kernel<E_DIM, R_DIM, false><<<gHi, 256, 0, stream>>>(tmp,   Wq_hi, bq_hi, nullptr, qbuf, 0.125f);
  lr_gemm_kernel<R_DIM, E_DIM, true ><<<gLo, 256, 0, stream>>>(key_,  Wk_lo, bk_lo, tmp, nullptr, 1.0f);
  lr_gemm_kernel<E_DIM, R_DIM, false><<<gHi, 256, 0, stream>>>(tmp,   Wk_hi, bk_hi, nullptr, kbuf, 1.0f);
  lr_gemm_kernel<R_DIM, E_DIM, true ><<<gLo, 256, 0, stream>>>(value, Wv_lo, bv_lo, tmp, nullptr, 1.0f);
  lr_gemm_kernel<E_DIM, R_DIM, false><<<gHi, 256, 0, stream>>>(tmp,   Wv_hi, bv_hi, nullptr, vbuf, 1.0f);

  lr_transpose_v<<<(B_DIM * H_NUM * HD_DIM * S_DIM) / 256, 256, 0, stream>>>(vbuf, vtb);

  lr_flash_attn<<<dim3(S_DIM / 64, B_DIM * H_NUM), 128, 0, stream>>>(qbuf, kbuf, vtb, attn);

  // output low-rank projection -> d_out (f32)
  lr_gemm_kernel<R_DIM, E_DIM, true ><<<gLo, 256, 0, stream>>>(attn, Wo_lo, bo_lo, tmp, nullptr, 1.0f);
  lr_gemm_kernel<E_DIM, R_DIM, true ><<<gHi, 256, 0, stream>>>(tmp,  Wo_hi, bo_hi, (float*)d_out, nullptr, 1.0f);
}
